// Int4Linear_29669634081275
// MI455X (gfx1250) — matrix operand order repaired and checked by reading, not compiled, not run on hardware
//
#include <hip/hip_runtime.h>

#define IN_F   4096
#define OUT_F  12288
#define GROUP  128

#define BM 128
#define BN 128
#define BK 64
#define STRIDE 44   // dwords per LDS row (88 bf16; padded: conflict-free & 16B aligned)

typedef __attribute__((ext_vector_type(16))) __bf16        v16bf;
typedef __attribute__((ext_vector_type(8)))  float         v8f;
typedef __attribute__((ext_vector_type(8)))  unsigned int  u32x8;
typedef __attribute__((ext_vector_type(2)))  float         f32x2;
typedef __attribute__((ext_vector_type(2)))  __bf16        bf16x2;

// packed f32x2 -> bf16x2 (RNE); lowers to v_cvt_pk_bf16_f32 on gfx1250
__device__ __forceinline__ unsigned int pack2_bf16(float lo, float hi) {
    f32x2 f;
    f.x = lo;
    f.y = hi;
    bf16x2 h = __builtin_convertvector(f, bf16x2);
    return __builtin_bit_cast(unsigned int, h);
}

__global__ __launch_bounds__(256)
void int4_linear_wmma_kernel(const float* __restrict__ x,
                             const int*   __restrict__ wp,
                             const float* __restrict__ scale,
                             const float* __restrict__ zp,
                             float* __restrict__ out)
{
    __shared__ unsigned int ldsA[BM * STRIDE];   // 128 rows x 64 bf16 (padded)
    __shared__ unsigned int ldsW[BN * STRIDE];   // 128 rows x 64 bf16 (padded)

    const int tid    = threadIdx.x;
    const int lane   = tid & 31;
    const int laneLo = lane & 15;
    const int laneHi = lane >> 4;
    const int wave   = tid >> 5;   // 0..7
    const int wm     = wave & 3;   // M quadrant (32 rows each)
    const int wn     = wave >> 2;  // N half (64 cols each)

    const int mTile = blockIdx.y * BM;
    const int nTile = blockIdx.x * BN;

    // loader role: each thread owns one row, half the K-chunk (32 elems)
    const int lrow  = tid >> 1;    // 0..127
    const int lhalf = tid & 1;

    const float* gx = x  + (size_t)(mTile + lrow) * IN_F      + lhalf * 32;
    const int*   gw = wp + (size_t)(nTile + lrow) * (IN_F/2)  + lhalf * 16;
    const float* gs = scale + (size_t)(nTile + lrow) * (IN_F/GROUP);
    const float* gz = zp    + (size_t)(nTile + lrow) * (IN_F/GROUP);

    unsigned int* dstA = &ldsA[lrow * STRIDE + lhalf * 16];
    unsigned int* dstW = &ldsW[lrow * STRIDE + lhalf * 16];

    v8f acc[2][4];
    #pragma unroll
    for (int i = 0; i < 2; ++i)
        #pragma unroll
        for (int j = 0; j < 4; ++j)
            acc[i][j] = (v8f)0.0f;

    // ---- register double-buffer: raw chunk data lives in VGPRs while the
    //      previous chunk is being multiplied out of LDS ----
    float4 xr[8];
    int4   wr[4];
    float  sReg, zReg;

    {   // prologue: preload chunk 0
        const float4* xa = (const float4*)(gx + 0);
        #pragma unroll
        for (int i = 0; i < 8; ++i) xr[i] = xa[i];
        const int4* wa = (const int4*)(gw + 0);
        #pragma unroll
        for (int i = 0; i < 4; ++i) wr[i] = wa[i];
        sReg = gs[0];
        zReg = gz[0];
    }

    for (int k0 = 0; k0 < IN_F; k0 += BK) {
        // ---- convert preloaded X regs: f32 -> bf16 (packed cvt) -> LDS ----
        {
            unsigned int abuf[16];
            #pragma unroll
            for (int i = 0; i < 8; ++i) {
                float4 v = xr[i];
                abuf[2 * i]     = pack2_bf16(v.x, v.y);
                abuf[2 * i + 1] = pack2_bf16(v.z, v.w);
            }
            #pragma unroll
            for (int i = 0; i < 4; ++i) {
                uint4 r;
                r.x = abuf[4 * i];     r.y = abuf[4 * i + 1];
                r.z = abuf[4 * i + 2]; r.w = abuf[4 * i + 3];
                ((uint4*)dstA)[i] = r;
            }
        }

        // ---- dequant preloaded W regs: int4 nibbles -> bf16 -> LDS ----
        {
            const float s = sReg;
            const float z = zReg;
            unsigned int wbuf[16];
            #pragma unroll
            for (int i = 0; i < 4; ++i) {
                int vv[4] = {wr[i].x, wr[i].y, wr[i].z, wr[i].w};
                #pragma unroll
                for (int j = 0; j < 4; ++j) {
                    int b = vv[j];
                    float we = (float)((b >> 4) & 15) * s + z;  // even k (high nibble)
                    float wo = (float)( b       & 15) * s + z;  // odd  k (low nibble)
                    wbuf[i * 4 + j] = pack2_bf16(we, wo);
                }
            }
            #pragma unroll
            for (int i = 0; i < 4; ++i) {
                uint4 r;
                r.x = wbuf[4 * i];     r.y = wbuf[4 * i + 1];
                r.z = wbuf[4 * i + 2]; r.w = wbuf[4 * i + 3];
                ((uint4*)dstW)[i] = r;
            }
        }

        __syncthreads();

        // ---- issue next chunk's global loads; results not consumed until the
        //      next iteration's convert, so they overlap the WMMA block below ----
        {
            const int knext = (k0 + BK < IN_F) ? (k0 + BK) : k0;  // branch-free tail
            const float4* xa = (const float4*)(gx + knext);
            #pragma unroll
            for (int i = 0; i < 8; ++i) xr[i] = xa[i];
            const int4* wa = (const int4*)(gw + (knext >> 1));
            #pragma unroll
            for (int i = 0; i < 4; ++i) wr[i] = wa[i];
            const int g = knext >> 7;
            sReg = gs[g];
            zReg = gz[g];
        }

        // ---- WMMA: two K=32 steps over the staged 64-wide chunk ----
        #pragma unroll
        for (int km = 0; km < 2; ++km) {
            v16bf aFrag[2];
            v16bf bFrag[4];

            #pragma unroll
            for (int i = 0; i < 2; ++i) {
                // A 16x32 bf16: lane(0-15)=row M, holds K=[hi*8,+8) and K=[16+hi*8,+8)
                int row = wm * 32 + i * 16 + laneLo;
                const uint4* p0 = (const uint4*)&ldsA[row * STRIDE + km * 16 + laneHi * 4];
                const uint4* p1 = (const uint4*)&ldsA[row * STRIDE + km * 16 + 8 + laneHi * 4];
                uint4 a0 = *p0, a1 = *p1;
                u32x8 t;
                t[0] = a0.x; t[1] = a0.y; t[2] = a0.z; t[3] = a0.w;
                t[4] = a1.x; t[5] = a1.y; t[6] = a1.z; t[7] = a1.w;
                aFrag[i] = __builtin_bit_cast(v16bf, t);
            }

            #pragma unroll
            for (int j = 0; j < 4; ++j) {
                // B 32x16 bf16: lane holds column N=laneLo, K=[hi*8,+8) and K=[16+hi*8,+8)
                int col = wn * 64 + j * 16 + laneLo;
                const uint4* p = (const uint4*)&ldsW[col * STRIDE + km * 16 + laneHi * 4];
                uint4 b0 = p[0], b1 = p[2];
                u32x8 t;
                t[0] = b0.x; t[1] = b0.y; t[2] = b0.z; t[3] = b0.w;
                t[4] = b1.x; t[5] = b1.y; t[6] = b1.z; t[7] = b1.w;
                bFrag[j] = __builtin_bit_cast(v16bf, t);
            }

            #pragma unroll
            for (int i = 0; i < 2; ++i)
                #pragma unroll
                for (int j = 0; j < 4; ++j)
                    acc[i][j] = __builtin_amdgcn_wmma_f32_16x16x32_bf16(
                        false, aFrag[i], false, bFrag[j],
                        (short)0, acc[i][j], false, false);
        }

        __syncthreads();
    }

    // ---- epilogue: D layout -> C[m][n]; VGPR r holds M = laneHi*8 + r, N = laneLo ----
    #pragma unroll
    for (int i = 0; i < 2; ++i) {
        #pragma unroll
        for (int j = 0; j < 4; ++j) {
            int n = nTile + wn * 64 + j * 16 + laneLo;
            #pragma unroll
            for (int r = 0; r < 8; ++r) {
                int m = mTile + wm * 32 + i * 16 + laneHi * 8 + r;
                out[(size_t)m * OUT_F + n] = acc[i][j][r];
            }
        }
    }
}

extern "C" void kernel_launch(void* const* d_in, const int* in_sizes, int n_in,
                              void* d_out, int out_size, void* d_ws, size_t ws_size,
                              hipStream_t stream) {
    const float* x     = (const float*)d_in[0];
    const int*   wp    = (const int*)  d_in[1];
    const float* scale = (const float*)d_in[2];
    const float* zp    = (const float*)d_in[3];
    float*       out   = (float*)d_out;

    const int M = in_sizes[0] / IN_F;          // 4 * 2048 = 8192
    dim3 grid(OUT_F / BN, M / BM);             // (96, 64)
    dim3 block(256);
    int4_linear_wmma_kernel<<<grid, block, 0, stream>>>(x, wp, scale, zp, out);
}
